// MultiHeadSelfAttention_61495341744167
// MI455X (gfx1250) — matrix operand order repaired and checked by reading, not compiled, not run on hardware
//
#include <hip/hip_runtime.h>
#include <hip/hip_bf16.h>
#include <math.h>

#define B_  4
#define S_  2048
#define D_  768
#define H_  12
#define DH_ 64
#define BS_ (B_ * S_)

typedef __attribute__((ext_vector_type(16))) _Float16 v16h;
typedef __attribute__((ext_vector_type(8)))  _Float16 v8h;
typedef __attribute__((ext_vector_type(8)))  float    v8f;
typedef __attribute__((ext_vector_type(4)))  float    v4f;
typedef _Float16 half_t;

__device__ __forceinline__ v8f wmma_f16(v16h a, v16h b, v8f c) {
  // D = A(16x32 f16) * B(32x16 f16) + C(16x16 f32)
  return __builtin_amdgcn_wmma_f32_16x16x32_f16(
      /*neg_a=*/false, a, /*neg_b=*/false, b,
      /*c_mod=*/(short)0, c, /*reuse_a=*/false, /*reuse_b=*/false);
}

// ---- Fragment loaders (f16 only; conversion is hoisted to a pack pass) -----
// A (16x32 f16): lane holds row m = lane&15.
// element j -> k = (lane>>4)*8 + (j<8 ? j : j+8)   (caller passes kb8 = (lane>>4)*8 + kstep)
__device__ __forceinline__ v16h fragA_f16(const half_t* row, int kb8) {
  v8h lo = *(const v8h*)(row + kb8);
  v8h hi = *(const v8h*)(row + kb8 + 16);
  v16h f;
#pragma unroll
  for (int j = 0; j < 8; ++j) { f[j] = lo[j]; f[j + 8] = hi[j]; }
  return f;
}

// B (32x16 f16): lane holds column n = lane&15.
// element j -> k = (lane>>4)*8 + (j<8 ? j : j+8)   (caller passes kb16 = (lane>>4)*8 + kstep)
__device__ __forceinline__ v16h fragB_f16(const half_t* row, int kb16) {
  v8h lo = *(const v8h*)(row + kb16);
  v8h hi = *(const v8h*)(row + kb16 + 16);
  v16h f;
#pragma unroll
  for (int j = 0; j < 8; ++j) { f[j] = lo[j]; f[j + 8] = hi[j]; }
  return f;
}

// ---- Kernel 0: f32 -> f16 pack (bandwidth-bound, runs once) ----------------
__global__ __launch_bounds__(256) void pack_f16_kernel(
    const float* __restrict__ src, half_t* __restrict__ dst, int n)
{
  int i = (blockIdx.x * blockDim.x + threadIdx.x) * 8;
  if (i >= n) return;
  v4f a = *(const v4f*)(src + i);
  v4f b = *(const v4f*)(src + i + 4);
  v8h o;
#pragma unroll
  for (int j = 0; j < 4; ++j) {
    o[j]     = (_Float16)a[j];
    o[j + 4] = (_Float16)b[j];
  }
  *(v8h*)(dst + i) = o;
}

// ---- Kernel 1: fused QKV projection (f16 GEMM, 32x64 tile per wave) --------
// One wave: 32x64 tile of one of {Q,K,V} = X @ W^T + b.
// Q scaled by 1/sqrt(DH); V stored transposed (B,H,DH,S).
__global__ __launch_bounds__(256) void qkv_proj_kernel(
    const half_t* __restrict__ xq, const half_t* __restrict__ xk, const half_t* __restrict__ xv,
    const half_t* __restrict__ Wq, const float* __restrict__ bq,
    const half_t* __restrict__ Wk, const float* __restrict__ bk,
    const half_t* __restrict__ Wv, const float* __restrict__ bv,
    half_t* __restrict__ Qh, half_t* __restrict__ Kh, half_t* __restrict__ Vt)
{
  const int RT = BS_ / 32;   // 256 row tiles
  const int CT = D_ / 64;    // 12 col tiles
  int wave  = blockIdx.x * (blockDim.x >> 5) + (threadIdx.x >> 5);
  int which = wave / (RT * CT);            // 0=Q 1=K 2=V
  int rem   = wave - which * (RT * CT);
  int rt    = rem / CT;
  int ct    = rem - rt * CT;

  const half_t* X    = (which == 0) ? xq : (which == 1) ? xk : xv;
  const half_t* W    = (which == 0) ? Wq : (which == 1) ? Wk : Wv;
  const float*  bias = (which == 0) ? bq : (which == 1) ? bk : bv;

  int lane = threadIdx.x & 31;
  int n = lane & 15;
  int hiHalf = lane >> 4;
  int kb8 = hiHalf * 8, kb16 = hiHalf * 8;
  int row0 = rt * 32, col0 = ct * 64;

  const half_t* a0 = X + (size_t)(row0 + n) * D_;
  const half_t* a1 = X + (size_t)(row0 + 16 + n) * D_;
  const half_t* b0 = W + (size_t)(col0 + n) * D_;       // y = x W^T -> B col n = W row
  const half_t* b1 = W + (size_t)(col0 + 16 + n) * D_;
  const half_t* b2 = W + (size_t)(col0 + 32 + n) * D_;
  const half_t* b3 = W + (size_t)(col0 + 48 + n) * D_;

  v8f acc[2][4] = {};
#pragma unroll 2
  for (int k = 0; k < D_; k += 32) {
    v16h A0 = fragA_f16(a0, k + kb8);
    v16h A1 = fragA_f16(a1, k + kb8);
    v16h B0 = fragB_f16(b0, k + kb16);
    v16h B1 = fragB_f16(b1, k + kb16);
    v16h B2 = fragB_f16(b2, k + kb16);
    v16h B3 = fragB_f16(b3, k + kb16);
    acc[0][0] = wmma_f16(A0, B0, acc[0][0]);
    acc[0][1] = wmma_f16(A0, B1, acc[0][1]);
    acc[0][2] = wmma_f16(A0, B2, acc[0][2]);
    acc[0][3] = wmma_f16(A0, B3, acc[0][3]);
    acc[1][0] = wmma_f16(A1, B0, acc[1][0]);
    acc[1][1] = wmma_f16(A1, B1, acc[1][1]);
    acc[1][2] = wmma_f16(A1, B2, acc[1][2]);
    acc[1][3] = wmma_f16(A1, B3, acc[1][3]);
  }

#pragma unroll
  for (int j = 0; j < 4; ++j) {
    int gcol = col0 + j * 16 + n;
    float bn = bias[gcol];
    int h  = gcol >> 6;          // / DH_
    int dh = gcol & (DH_ - 1);
#pragma unroll
    for (int i = 0; i < 2; ++i) {
#pragma unroll
      for (int r = 0; r < 8; ++r) {
        int grow = row0 + i * 16 + r + hiHalf * 8;
        int bb = grow >> 11;            // / S_
        int s  = grow & (S_ - 1);
        float val = acc[i][j][r] + bn;
        if (which == 0) {
          Qh[(((size_t)bb * H_ + h) * S_ + s) * DH_ + dh] = (half_t)(val * 0.125f);
        } else if (which == 1) {
          Kh[(((size_t)bb * H_ + h) * S_ + s) * DH_ + dh] = (half_t)val;
        } else {
          Vt[(((size_t)bb * H_ + h) * DH_ + dh) * S_ + s] = (half_t)val;
        }
      }
    }
  }
}

// ---- Kernel 2: flash attention (one wave = 16 queries of one (b,h)) --------
__global__ __launch_bounds__(256) void attn_kernel(
    const half_t* __restrict__ Qh, const half_t* __restrict__ Kh,
    const half_t* __restrict__ Vt, const int* __restrict__ mask,
    half_t* __restrict__ ctx)
{
  __shared__ __align__(16) half_t Plds[8][16 * 32];   // 1KB per wave
  int wid  = threadIdx.x >> 5;
  int lane = threadIdx.x & 31;
  int wave = blockIdx.x * 8 + wid;

  int qt = wave & (S_ / 16 - 1);     // 0..127
  int bh = wave >> 7;                // 0..47
  int b  = bh / H_;
  int h  = bh - b * H_;

  const half_t* qbase = Qh + (size_t)bh * S_ * DH_;
  const half_t* kbase = Kh + (size_t)bh * S_ * DH_;
  const half_t* vbase = Vt + (size_t)bh * DH_ * S_;
  const int*    mrow  = mask + (size_t)b * S_;

  int n = lane & 15;
  int hiHalf = lane >> 4;
  int kb8 = hiHalf * 8, kb16 = hiHalf * 8;
  int q0 = qt * 16;

  // Q fragments for the whole head dim (K=64 -> two 16x32 A fragments)
  const half_t* qrow = qbase + (size_t)(q0 + n) * DH_;
  v16h qa0 = fragA_f16(qrow, kb8);
  v16h qa1 = fragA_f16(qrow, 32 + kb8);

  float mrun[8], lrun[8], corr[8];
  v8f O0 = {}, O1 = {}, O2 = {}, O3 = {};
#pragma unroll
  for (int r = 0; r < 8; ++r) { mrun[r] = -1e30f; lrun[r] = 0.f; }

  half_t* P = &Plds[wid][0];

  for (int k0 = 0; k0 < S_; k0 += 32) {
    // scores for 32 keys: two 16x16 tiles, K-dim = 64
    v8f s0 = {}, s1 = {};
    {
      const half_t* kr0 = kbase + (size_t)(k0 + n) * DH_;
      s0 = wmma_f16(qa0, fragB_f16(kr0, kb16), s0);
      s0 = wmma_f16(qa1, fragB_f16(kr0, 32 + kb16), s0);
      const half_t* kr1 = kbase + (size_t)(k0 + 16 + n) * DH_;
      s1 = wmma_f16(qa0, fragB_f16(kr1, kb16), s1);
      s1 = wmma_f16(qa1, fragB_f16(kr1, 32 + kb16), s1);
    }
    bool ok0 = mrow[k0 + n] != 0;
    bool ok1 = mrow[k0 + 16 + n] != 0;

    // online softmax: rows live across 16-lane halves; xor 1/2/4/8 stays in-group
#pragma unroll
    for (int r = 0; r < 8; ++r) {
      float a = ok0 ? s0[r] : -1e30f;
      float c = ok1 ? s1[r] : -1e30f;
      float v = fmaxf(a, c);
#pragma unroll
      for (int off = 1; off < 16; off <<= 1)
        v = fmaxf(v, __shfl_xor(v, off, 32));
      float mn = fmaxf(mrun[r], v);
      float cr = __expf(mrun[r] - mn);
      mrun[r] = mn;
      corr[r] = cr;
      float p0 = __expf(a - mn);
      float p1 = __expf(c - mn);
      s0[r] = p0; s1[r] = p1;
      float rs = p0 + p1;
#pragma unroll
      for (int off = 1; off < 16; off <<= 1)
        rs += __shfl_xor(rs, off, 32);
      lrun[r] = lrun[r] * cr + rs;
    }
#pragma unroll
    for (int r = 0; r < 8; ++r) {
      O0[r] *= corr[r]; O1[r] *= corr[r]; O2[r] *= corr[r]; O3[r] *= corr[r];
    }

    // C-layout -> A-layout via per-wave LDS tile
#pragma unroll
    for (int r = 0; r < 8; ++r) {
      int mm = r + hiHalf * 8;
      P[mm * 32 + n]      = (half_t)s0[r];
      P[mm * 32 + 16 + n] = (half_t)s1[r];
    }
    asm volatile("s_wait_dscnt 0" ::: "memory");
    v16h pa = fragA_f16(P + n * 32, kb8);

    // O += P(16x32) @ V(32x64): V^T rows are contiguous over keys
    O0 = wmma_f16(pa, fragB_f16(vbase + (size_t)(0  + n) * S_ + k0, kb16), O0);
    O1 = wmma_f16(pa, fragB_f16(vbase + (size_t)(16 + n) * S_ + k0, kb16), O1);
    O2 = wmma_f16(pa, fragB_f16(vbase + (size_t)(32 + n) * S_ + k0, kb16), O2);
    O3 = wmma_f16(pa, fragB_f16(vbase + (size_t)(48 + n) * S_ + k0, kb16), O3);
  }

  // finalize and write context in (B,S,D) f16 layout for the output GEMM
#pragma unroll
  for (int r = 0; r < 8; ++r) {
    float inv = 1.0f / lrun[r];
    int mm = r + hiHalf * 8;
    size_t base = ((size_t)b * S_ + (q0 + mm)) * D_ + (size_t)h * DH_ + n;
    ctx[base + 0]  = (half_t)(O0[r] * inv);
    ctx[base + 16] = (half_t)(O1[r] * inv);
    ctx[base + 32] = (half_t)(O2[r] * inv);
    ctx[base + 48] = (half_t)(O3[r] * inv);
  }
}

// ---- Kernel 3: output projection: out = ctx @ Wo^T + bo (32x64/wave) -------
__global__ __launch_bounds__(256) void out_proj_kernel(
    const half_t* __restrict__ ctx, const half_t* __restrict__ Wo,
    const float* __restrict__ bo, float* __restrict__ out)
{
  const int CT = D_ / 64;    // 12
  int wave = blockIdx.x * (blockDim.x >> 5) + (threadIdx.x >> 5);
  int rt = wave / CT;
  int ct = wave - rt * CT;

  int lane = threadIdx.x & 31;
  int n = lane & 15;
  int hiHalf = lane >> 4;
  int kb8 = hiHalf * 8, kb16 = hiHalf * 8;
  int row0 = rt * 32, col0 = ct * 64;

  const half_t* a0 = ctx + (size_t)(row0 + n) * D_;
  const half_t* a1 = ctx + (size_t)(row0 + 16 + n) * D_;
  const half_t* b0 = Wo + (size_t)(col0 + n) * D_;
  const half_t* b1 = Wo + (size_t)(col0 + 16 + n) * D_;
  const half_t* b2 = Wo + (size_t)(col0 + 32 + n) * D_;
  const half_t* b3 = Wo + (size_t)(col0 + 48 + n) * D_;

  v8f acc[2][4] = {};
#pragma unroll 2
  for (int k = 0; k < D_; k += 32) {
    v16h A0 = fragA_f16(a0, k + kb8);
    v16h A1 = fragA_f16(a1, k + kb8);
    v16h B0 = fragB_f16(b0, k + kb16);
    v16h B1 = fragB_f16(b1, k + kb16);
    v16h B2 = fragB_f16(b2, k + kb16);
    v16h B3 = fragB_f16(b3, k + kb16);
    acc[0][0] = wmma_f16(A0, B0, acc[0][0]);
    acc[0][1] = wmma_f16(A0, B1, acc[0][1]);
    acc[0][2] = wmma_f16(A0, B2, acc[0][2]);
    acc[0][3] = wmma_f16(A0, B3, acc[0][3]);
    acc[1][0] = wmma_f16(A1, B0, acc[1][0]);
    acc[1][1] = wmma_f16(A1, B1, acc[1][1]);
    acc[1][2] = wmma_f16(A1, B2, acc[1][2]);
    acc[1][3] = wmma_f16(A1, B3, acc[1][3]);
  }

#pragma unroll
  for (int j = 0; j < 4; ++j) {
    int gcol = col0 + j * 16 + n;
    float bn = bo[gcol];
#pragma unroll
    for (int i = 0; i < 2; ++i) {
#pragma unroll
      for (int r = 0; r < 8; ++r) {
        int grow = row0 + i * 16 + r + hiHalf * 8;
        out[(size_t)grow * D_ + gcol] = acc[i][j][r] + bn;
      }
    }
  }
}

// ---- Host launcher ---------------------------------------------------------
extern "C" void kernel_launch(void* const* d_in, const int* in_sizes, int n_in,
                              void* d_out, int out_size, void* d_ws, size_t ws_size,
                              hipStream_t stream) {
  const float* query = (const float*)d_in[0];
  const float* key   = (const float*)d_in[1];
  const float* value = (const float*)d_in[2];
  const int*   mask  = (const int*)d_in[3];
  const float* Wq = (const float*)d_in[4];
  const float* bq = (const float*)d_in[5];
  const float* Wk = (const float*)d_in[6];
  const float* bk = (const float*)d_in[7];
  const float* Wv = (const float*)d_in[8];
  const float* bv = (const float*)d_in[9];
  const float* Wo = (const float*)d_in[10];
  const float* bo = (const float*)d_in[11];
  float* out = (float*)d_out;
  (void)in_sizes; (void)n_in; (void)out_size; (void)ws_size;

  const size_t NX = (size_t)BS_ * D_;   // 6,291,456 elements
  const size_t NW = (size_t)D_ * D_;    //   589,824 elements
  half_t* p = (half_t*)d_ws;
  half_t* Xq  = p; p += NX;            // f16 copies of inputs
  half_t* Xk  = p; p += NX;
  half_t* Xv  = p; p += NX;
  half_t* Wqh = p; p += NW;            // f16 copies of weights
  half_t* Wkh = p; p += NW;
  half_t* Wvh = p; p += NW;
  half_t* Woh = p; p += NW;
  half_t* Qh  = p; p += NX;            // (B,H,S,DH), pre-scaled by 1/sqrt(DH)
  half_t* Kh  = p; p += NX;            // (B,H,S,DH)
  half_t* Vt  = p; p += NX;            // (B,H,DH,S) transposed
  half_t* ctx = p; p += NX;            // (B,S,D)

  dim3 blk(256);
  // pack pass: fp32 -> f16 once (removes cvt from GEMM hot loops)
  pack_f16_kernel<<<dim3((int)(NX / 8 / 256)), blk, 0, stream>>>(query, Xq, (int)NX);
  pack_f16_kernel<<<dim3((int)(NX / 8 / 256)), blk, 0, stream>>>(key,   Xk, (int)NX);
  pack_f16_kernel<<<dim3((int)(NX / 8 / 256)), blk, 0, stream>>>(value, Xv, (int)NX);
  pack_f16_kernel<<<dim3((int)(NW / 8 / 256)), blk, 0, stream>>>(Wq, Wqh, (int)NW);
  pack_f16_kernel<<<dim3((int)(NW / 8 / 256)), blk, 0, stream>>>(Wk, Wkh, (int)NW);
  pack_f16_kernel<<<dim3((int)(NW / 8 / 256)), blk, 0, stream>>>(Wv, Wvh, (int)NW);
  pack_f16_kernel<<<dim3((int)(NW / 8 / 256)), blk, 0, stream>>>(Wo, Woh, (int)NW);

  // 3 matrices * 256 row-tiles * 12 col-tiles / 8 waves per block = 1152
  qkv_proj_kernel<<<dim3(3 * (BS_ / 32) * (D_ / 64) / 8), blk, 0, stream>>>(
      Xq, Xk, Xv, Wqh, bq, Wkh, bk, Wvh, bv, Qh, Kh, Vt);
  // B*H*(S/16) = 6144 waves / 8 = 768 blocks
  attn_kernel<<<dim3(B_ * H_ * (S_ / 16) / 8), blk, 0, stream>>>(Qh, Kh, Vt, mask, ctx);
  // 256 * 12 / 8 = 384 blocks
  out_proj_kernel<<<dim3((BS_ / 32) * (D_ / 64) / 8), blk, 0, stream>>>(ctx, Woh, bo, out);
}
